// GATLayer_50311246905882
// MI455X (gfx1250) — compile-verified, matrix operand order checked by reading, not run on hardware
//
#include <hip/hip_runtime.h>
#include <hip/hip_bf16.h>
#include <math.h>

#define N_NODES 50000
#define E_EDGES 800000
#define IN_F    256
#define HID     128
#define HEADS1  8
#define OUT_F   64
#define NEG_SLOPE 0.2f
#define EPS_A   1e-16f
#define ETOT    (E_EDGES + N_NODES)

typedef __attribute__((ext_vector_type(16))) __bf16 v16bf;
typedef __attribute__((ext_vector_type(8)))  float  v8f;

// ---------------------------------------------------------------------------
// float atomic max via signed/unsigned integer ordering trick
// (lowers to global_atomic_max_i32 / global_atomic_min_u32)
// ---------------------------------------------------------------------------
__device__ __forceinline__ void atomic_max_f32(float* addr, float v) {
    if (v >= 0.0f) atomicMax((int*)addr, __float_as_int(v));
    else           atomicMin((unsigned int*)addr, (unsigned int)__float_as_int(v));
}

__global__ void fill_kernel(float* __restrict__ p, float v, int count) {
    int i = blockIdx.x * blockDim.x + threadIdx.x;
    if (i < count) p[i] = v;
}

// ---------------------------------------------------------------------------
// GEMM1: h = x[50000,256] @ W1[256,128]  (bf16 WMMA, f32 accum)
// block = 256 threads = 8 waves; wave w owns 16-col tile == head w.
// Also computes a_s[n,h] = <h[n,h,:], att_src1[h,:]> (same for a_d) via
// shfl reduction straight from the WMMA accumulator.
// ---------------------------------------------------------------------------
__global__ __launch_bounds__(256) void gemm1_kernel(
    const float* __restrict__ x, const float* __restrict__ W,
    const float* __restrict__ att_s, const float* __restrict__ att_d,
    float* __restrict__ h, float* __restrict__ as_, float* __restrict__ ad_)
{
    const int wave = threadIdx.x >> 5;      // 0..7 : column tile / head
    const int lane = threadIdx.x & 31;
    const int half = lane >> 4;             // K-half selector (A/B layout)
    const int q    = lane & 15;             // M for A, N for B/D
    const int row0 = blockIdx.x * 16;       // 3125 blocks * 16 = 50000 exact
    const int m    = row0 + q;
    const int n    = wave * 16 + q;

    v8f c = {};
    for (int kb = 0; kb < IN_F; kb += 32) {
        v16bf a, b;
#pragma unroll
        for (int v = 0; v < 8; ++v) {
            const int k0 = (v < 4) ? (2 * v + 8 * half)
                                   : (16 + 2 * (v - 4) + 8 * half);
            const int k  = kb + k0;
            const float2 av = *(const float2*)(x + (size_t)m * IN_F + k);
            a[2 * v]     = (__bf16)av.x;
            a[2 * v + 1] = (__bf16)av.y;
            b[2 * v]     = (__bf16)W[(size_t)k       * HID + n];
            b[2 * v + 1] = (__bf16)W[(size_t)(k + 1) * HID + n];
        }
        c = __builtin_amdgcn_wmma_f32_16x16x32_bf16(false, a, false, b,
                                                    (short)0, c, false, false);
    }

    const float atts = att_s[wave * 16 + q];
    const float attd = att_d[wave * 16 + q];
#pragma unroll
    for (int r = 0; r < 8; ++r) {
        const int row = row0 + r + 8 * half;
        h[(size_t)row * HID + wave * 16 + q] = c[r];
        float s = c[r] * atts, d = c[r] * attd;
#pragma unroll
        for (int msk = 8; msk >= 1; msk >>= 1) {
            s += __shfl_xor(s, msk, 32);
            d += __shfl_xor(d, msk, 32);
        }
        if (q == 0) {
            as_[(size_t)row * HEADS1 + wave] = s;
            ad_[(size_t)row * HEADS1 + wave] = d;
        }
    }
}

// ---------------------------------------------------------------------------
// GEMM2: g = h2[50000,128] @ W2[128,64]  (bf16 WMMA, f32 accum)
// block = 128 threads = 4 waves (one per 16-col tile).
// ---------------------------------------------------------------------------
__global__ __launch_bounds__(128) void gemm2_kernel(
    const float* __restrict__ h2, const float* __restrict__ W,
    float* __restrict__ g)
{
    const int wave = threadIdx.x >> 5;      // 0..3
    const int lane = threadIdx.x & 31;
    const int half = lane >> 4;
    const int q    = lane & 15;
    const int row0 = blockIdx.x * 16;
    const int m    = row0 + q;
    const int n    = wave * 16 + q;

    v8f c = {};
    for (int kb = 0; kb < HID; kb += 32) {
        v16bf a, b;
#pragma unroll
        for (int v = 0; v < 8; ++v) {
            const int k0 = (v < 4) ? (2 * v + 8 * half)
                                   : (16 + 2 * (v - 4) + 8 * half);
            const int k  = kb + k0;
            const float2 av = *(const float2*)(h2 + (size_t)m * HID + k);
            a[2 * v]     = (__bf16)av.x;
            a[2 * v + 1] = (__bf16)av.y;
            b[2 * v]     = (__bf16)W[(size_t)k       * OUT_F + n];
            b[2 * v + 1] = (__bf16)W[(size_t)(k + 1) * OUT_F + n];
        }
        c = __builtin_amdgcn_wmma_f32_16x16x32_bf16(false, a, false, b,
                                                    (short)0, c, false, false);
    }
#pragma unroll
    for (int r = 0; r < 8; ++r) {
        const int row = row0 + r + 8 * half;
        g[(size_t)row * OUT_F + wave * 16 + q] = c[r];
    }
}

// a_s2[n] = <g[n,:], att_src2>, a_d2 likewise (1 head, 64 ch)
__global__ void att2_kernel(const float* __restrict__ g,
                            const float* __restrict__ att_s,
                            const float* __restrict__ att_d,
                            float* __restrict__ as2, float* __restrict__ ad2)
{
    int i = blockIdx.x * blockDim.x + threadIdx.x;
    if (i >= N_NODES) return;
    float s = 0.f, d = 0.f;
    for (int j = 0; j < OUT_F; ++j) {
        float v = g[(size_t)i * OUT_F + j];
        s += v * att_s[j];
        d += v * att_d[j];
    }
    as2[i] = s;
    ad2[i] = d;
}

// ---------------------------------------------------------------------------
// Edge passes.  Self-loops are implicit: eid >= E -> src = dst = eid - E.
// edges: d_in[1] flat [2,E]; src = edges[0:E], dst = edges[E:2E].
// ---------------------------------------------------------------------------
__device__ __forceinline__ void edge_sd(const int* __restrict__ edges,
                                        int eid, int& s, int& d)
{
    if (eid < E_EDGES) { s = edges[eid]; d = edges[E_EDGES + eid]; }
    else               { s = eid - E_EDGES; d = s; }
}

__global__ void edge_max_kernel(const int* __restrict__ edges, int H,
                                const float* __restrict__ as_,
                                const float* __restrict__ ad_,
                                float* __restrict__ emax)
{
    int tid = blockIdx.x * blockDim.x + threadIdx.x;
    if (tid >= ETOT * H) return;
    const int eid = tid / H, hh = tid - eid * H;
    int s, d; edge_sd(edges, eid, s, d);
    float e = as_[(size_t)s * H + hh] + ad_[(size_t)d * H + hh];
    e = (e > 0.f) ? e : NEG_SLOPE * e;
    atomic_max_f32(&emax[(size_t)d * H + hh], e);
}

__global__ void edge_expsum_kernel(const int* __restrict__ edges, int H,
                                   const float* __restrict__ as_,
                                   const float* __restrict__ ad_,
                                   const float* __restrict__ emax,
                                   float* __restrict__ denom)
{
    int tid = blockIdx.x * blockDim.x + threadIdx.x;
    if (tid >= ETOT * H) return;
    const int eid = tid / H, hh = tid - eid * H;
    int s, d; edge_sd(edges, eid, s, d);
    float e = as_[(size_t)s * H + hh] + ad_[(size_t)d * H + hh];
    e = (e > 0.f) ? e : NEG_SLOPE * e;
    float w = __expf(e - emax[(size_t)d * H + hh]);
    atomicAdd(&denom[(size_t)d * H + hh], w);
}

// scatter: agg[dst, h*C + c] += msg[src, h*C + c] * alpha.
// threads = ETOT * H * chunks; each thread covers C/chunks channels.
__global__ void edge_scatter_kernel(const int* __restrict__ edges,
                                    int H, int C, int chunks,
                                    const float* __restrict__ as_,
                                    const float* __restrict__ ad_,
                                    const float* __restrict__ emax,
                                    const float* __restrict__ denom,
                                    const float* __restrict__ msg,
                                    float* __restrict__ agg)
{
    int tid = blockIdx.x * blockDim.x + threadIdx.x;
    const int per = H * chunks;
    if (tid >= ETOT * per) return;
    const int eid = tid / per;
    const int rem = tid - eid * per;
    const int hh  = rem / chunks;
    const int ch  = rem - hh * chunks;
    const int cw  = C / chunks;
    int s, d; edge_sd(edges, eid, s, d);
    float e = as_[(size_t)s * H + hh] + ad_[(size_t)d * H + hh];
    e = (e > 0.f) ? e : NEG_SLOPE * e;
    float w = __expf(e - emax[(size_t)d * H + hh]);
    float alpha = w / (denom[(size_t)d * H + hh] + EPS_A);
    const int HC = H * C;
    const float* srow = msg + (size_t)s * HC + hh * C + ch * cw;
    float*       drow = agg + (size_t)d * HC + hh * C + ch * cw;
    for (int c = 0; c < cw; ++c)
        atomicAdd(drow + c, srow[c] * alpha);
}

// h2 = elu(agg1 + b1) in place
__global__ void bias_elu_kernel(float* __restrict__ agg,
                                const float* __restrict__ bias, int count)
{
    int i = blockIdx.x * blockDim.x + threadIdx.x;
    if (i >= count) return;
    float v = agg[i] + bias[i % HID];
    agg[i] = (v > 0.f) ? v : (__expf(v) - 1.f);
}

// log_softmax over 64 channels, one warp per node (2 values per lane)
__global__ __launch_bounds__(256) void logsoftmax_kernel(
    const float* __restrict__ agg2, const float* __restrict__ b2,
    float* __restrict__ out)
{
    int warp = (blockIdx.x * blockDim.x + threadIdx.x) >> 5;
    int lane = threadIdx.x & 31;
    if (warp >= N_NODES) return;
    float v0 = agg2[(size_t)warp * OUT_F + lane]      + b2[lane];
    float v1 = agg2[(size_t)warp * OUT_F + 32 + lane] + b2[32 + lane];
    float mx = fmaxf(v0, v1);
#pragma unroll
    for (int msk = 16; msk >= 1; msk >>= 1)
        mx = fmaxf(mx, __shfl_xor(mx, msk, 32));
    float s = __expf(v0 - mx) + __expf(v1 - mx);
#pragma unroll
    for (int msk = 16; msk >= 1; msk >>= 1)
        s += __shfl_xor(s, msk, 32);
    float l = __logf(s) + mx;
    out[(size_t)warp * OUT_F + lane]      = v0 - l;
    out[(size_t)warp * OUT_F + 32 + lane] = v1 - l;
}

// ---------------------------------------------------------------------------
static inline int gridFor(long long threads, int block) {
    return (int)((threads + block - 1) / block);
}

extern "C" void kernel_launch(void* const* d_in, const int* in_sizes, int n_in,
                              void* d_out, int out_size, void* d_ws, size_t ws_size,
                              hipStream_t stream)
{
    const float* x        = (const float*)d_in[0];
    const int*   edges    = (const int*)  d_in[1];
    const float* W1       = (const float*)d_in[2];
    const float* att_src1 = (const float*)d_in[3];
    const float* att_dst1 = (const float*)d_in[4];
    const float* b1       = (const float*)d_in[5];
    const float* W2       = (const float*)d_in[6];
    const float* att_src2 = (const float*)d_in[7];
    const float* att_dst2 = (const float*)d_in[8];
    const float* b2       = (const float*)d_in[9];
    float*       out      = (float*)d_out;

    // workspace layout (floats), regions reused between layers
    float* f    = (float*)d_ws;
    float* h1   = f;                               // [N,128]  layer1 features / later g [N,64]
    float* agg  = f + (size_t)N_NODES * HID;       // [N,128]  agg1 -> h2 (in place) -> agg2 [N,64]
    float* aux  = agg + (size_t)N_NODES * HID;     // [N,32]
    float* as1   = aux;
    float* ad1   = aux + (size_t)N_NODES * HEADS1;
    float* emax1 = aux + (size_t)2 * N_NODES * HEADS1;
    float* den1  = aux + (size_t)3 * N_NODES * HEADS1;
    float* g     = h1;                 // layer-2 features (h1 dead after L1 scatter)
    float* as2   = aux;                // L1 att buffers dead after L1 scatter
    float* ad2   = aux + N_NODES;
    float* emax2 = aux + (size_t)2 * N_NODES;
    float* den2  = aux + (size_t)3 * N_NODES;
    float* agg2  = agg;                // h2 dead after gemm2

    const int BLK = 256;
    const float NEG_INF = -INFINITY;

    // ---- layer 1 ----
    gemm1_kernel<<<N_NODES / 16, 256, 0, stream>>>(x, W1, att_src1, att_dst1,
                                                   h1, as1, ad1);
    fill_kernel<<<gridFor((long long)N_NODES * HEADS1, BLK), BLK, 0, stream>>>(
        emax1, NEG_INF, N_NODES * HEADS1);
    fill_kernel<<<gridFor((long long)N_NODES * HEADS1, BLK), BLK, 0, stream>>>(
        den1, 0.f, N_NODES * HEADS1);
    fill_kernel<<<gridFor((long long)N_NODES * HID, BLK), BLK, 0, stream>>>(
        agg, 0.f, N_NODES * HID);

    long long t1 = (long long)ETOT * HEADS1;
    edge_max_kernel<<<gridFor(t1, BLK), BLK, 0, stream>>>(edges, HEADS1, as1, ad1, emax1);
    edge_expsum_kernel<<<gridFor(t1, BLK), BLK, 0, stream>>>(edges, HEADS1, as1, ad1,
                                                             emax1, den1);
    edge_scatter_kernel<<<gridFor(t1, BLK), BLK, 0, stream>>>(
        edges, HEADS1, /*C=*/16, /*chunks=*/1, as1, ad1, emax1, den1, h1, agg);

    bias_elu_kernel<<<gridFor((long long)N_NODES * HID, BLK), BLK, 0, stream>>>(
        agg, b1, N_NODES * HID);

    // ---- layer 2 ----
    gemm2_kernel<<<N_NODES / 16, 128, 0, stream>>>(agg, W2, g);
    att2_kernel<<<gridFor(N_NODES, BLK), BLK, 0, stream>>>(g, att_src2, att_dst2,
                                                           as2, ad2);
    fill_kernel<<<gridFor(N_NODES, BLK), BLK, 0, stream>>>(emax2, NEG_INF, N_NODES);
    fill_kernel<<<gridFor(N_NODES, BLK), BLK, 0, stream>>>(den2, 0.f, N_NODES);
    fill_kernel<<<gridFor((long long)N_NODES * OUT_F, BLK), BLK, 0, stream>>>(
        agg2, 0.f, N_NODES * OUT_F);

    long long t2 = (long long)ETOT;
    edge_max_kernel<<<gridFor(t2, BLK), BLK, 0, stream>>>(edges, 1, as2, ad2, emax2);
    edge_expsum_kernel<<<gridFor(t2, BLK), BLK, 0, stream>>>(edges, 1, as2, ad2,
                                                             emax2, den2);
    edge_scatter_kernel<<<gridFor(t2 * 4, BLK), BLK, 0, stream>>>(
        edges, 1, /*C=*/OUT_F, /*chunks=*/4, as2, ad2, emax2, den2, g, agg2);

    logsoftmax_kernel<<<gridFor((long long)N_NODES * 32, BLK), BLK, 0, stream>>>(
        agg2, b2, out);
}
